// ViTEncoder3D_26371099198041
// MI455X (gfx1250) — matrix operand order repaired and checked by reading, not compiled, not run on hardware
//
#include <hip/hip_runtime.h>

// ---------------------------------------------------------------------------
// Types
// ---------------------------------------------------------------------------
typedef __attribute__((ext_vector_type(16))) __bf16 v16bf;
typedef __attribute__((ext_vector_type(8)))  __bf16 v8bf;
typedef __attribute__((ext_vector_type(8)))  float  v8f;

#define BDIM   768
#define NTOK   512        // tokens per batch
#define MROWS  1024       // B * NTOK
#define QKVN   2304
#define MLPN   3072
#define NHEAD  12
#define DHEAD  64
#define NBH    24         // B * NHEAD
#define KPATCH 512        // CH * 8^3

__device__ inline __bf16 f2bf(float f) {
    unsigned u = __float_as_uint(f);
    unsigned r = (u + 0x7FFFu + ((u >> 16) & 1u)) >> 16;   // round-to-nearest-even
    unsigned short s = (unsigned short)r;
    __bf16 b;
    __builtin_memcpy(&b, &s, 2);
    return b;
}

__device__ inline float wave_sum(float v) {
#pragma unroll
    for (int m = 16; m >= 1; m >>= 1) v += __shfl_xor(v, m, 32);
    return v;
}
__device__ inline float wave_max(float v) {
#pragma unroll
    for (int m = 16; m >= 1; m >>= 1) v = fmaxf(v, __shfl_xor(v, m, 32));
    return v;
}

// ---------------------------------------------------------------------------
// Prep kernels
// ---------------------------------------------------------------------------
__global__ void k_convert(const float* __restrict__ src, __bf16* __restrict__ dst, int n) {
    int i = blockIdx.x * 256 + threadIdx.x;
    if (i < n) dst[i] = f2bf(src[i]);
}

// src f32 [R,C] row-major -> dst bf16 [C,R] row-major (i.e. transposed)
__global__ void k_transpose(const float* __restrict__ src, __bf16* __restrict__ dst, int R, int C) {
    int i = blockIdx.x * 256 + threadIdx.x;
    if (i >= R * C) return;
    int r = i / C, c = i % C;
    dst[(long long)c * R + r] = f2bf(src[i]);
}

// volume [2,1,64,64,64] f32 -> patch bf16 [1024, 512] (row = b*512 + token)
__global__ void k_im2col(const float* __restrict__ vol, __bf16* __restrict__ patch) {
    int i = blockIdx.x * 256 + threadIdx.x;
    if (i >= MROWS * KPATCH) return;
    int kk = i & 511;
    int row = i >> 9;
    int b = row >> 9, p = row & 511;
    int pd = p >> 6, ph = (p >> 3) & 7, pw = p & 7;
    int kd = kk >> 6, kh = (kk >> 3) & 7, kw = kk & 7;
    long long idx = (long long)b * 262144 + (long long)(pd * 8 + kd) * 4096
                  + (ph * 8 + kh) * 64 + (pw * 8 + kw);
    patch[i] = f2bf(vol[idx]);
}

// qkv bf16 [1024, 2304] -> q,k [BH,512,64] and vT [BH,64,512]
__global__ void k_qkv_split(const __bf16* __restrict__ qkv, __bf16* __restrict__ q,
                            __bf16* __restrict__ k, __bf16* __restrict__ vT) {
    int i = blockIdx.x * 256 + threadIdx.x;
    if (i >= MROWS * QKVN) return;
    int col = i % QKVN;
    int rt  = i / QKVN;
    int b = rt >> 9, n = rt & 511;
    int which = col / BDIM;
    int hc = col - which * BDIM;
    int h = hc >> 6, d = hc & 63;
    __bf16 val = qkv[i];
    int bh = b * NHEAD + h;
    if (which == 0)      q[((long long)bh * NTOK + n) * DHEAD + d] = val;
    else if (which == 1) k[((long long)bh * NTOK + n) * DHEAD + d] = val;
    else                 vT[((long long)bh * DHEAD + d) * NTOK + n] = val;
}

// ---------------------------------------------------------------------------
// LayerNorm: one wave per row
// ---------------------------------------------------------------------------
template <bool OUT_BF16>
__global__ void k_layernorm(const float* __restrict__ x, const float* __restrict__ w,
                            const float* __restrict__ b, void* __restrict__ out, int rows, int D) {
    int wave = threadIdx.x >> 5, lane = threadIdx.x & 31;
    int row = blockIdx.x * (blockDim.x >> 5) + wave;
    if (row >= rows) return;
    const float* xr = x + (long long)row * D;
    float s = 0.f;
    for (int i = lane; i < D; i += 32) s += xr[i];
    float mean = wave_sum(s) / D;
    float v = 0.f;
    for (int i = lane; i < D; i += 32) { float d = xr[i] - mean; v += d * d; }
    float rstd = rsqrtf(wave_sum(v) / D + 1e-5f);
    for (int i = lane; i < D; i += 32) {
        float o = (xr[i] - mean) * rstd * w[i] + b[i];
        if (OUT_BF16) ((__bf16*)out)[(long long)row * D + i] = f2bf(o);
        else          ((float*)out)[(long long)row * D + i]  = o;
    }
}

// ---------------------------------------------------------------------------
// Softmax: one wave per row of S [rows, n], write bf16 P
// ---------------------------------------------------------------------------
__global__ void k_softmax(const float* __restrict__ S, __bf16* __restrict__ P, int rows, int n) {
    int wave = threadIdx.x >> 5, lane = threadIdx.x & 31;
    int row = blockIdx.x * (blockDim.x >> 5) + wave;
    if (row >= rows) return;
    const float* sr = S + (long long)row * n;
    float m = -3.0e38f;
    for (int i = lane; i < n; i += 32) m = fmaxf(m, sr[i]);
    m = wave_max(m);
    float sum = 0.f;
    for (int i = lane; i < n; i += 32) sum += __expf(sr[i] - m);
    sum = wave_sum(sum);
    float inv = 1.0f / sum;
    for (int i = lane; i < n; i += 32)
        P[(long long)row * n + i] = f2bf(__expf(sr[i] - m) * inv);
}

// ---------------------------------------------------------------------------
// Register-blocked bf16 WMMA GEMM: C = act(alpha * A @ B + bias) [+ residual]
//   Each wave computes a 64x32 super-tile: 4 M-subtiles x 2 N-subtiles,
//   8 accumulators; per K-step of 32: 4 B-frag loads + 8 A-frag loads -> 8 WMMA.
//   A  : bf16 row-major [M,K]  (per batch z: + z*sA)
//   BT : bf16 row-major [N,K]  (B transposed; per batch z: + z*sB)
//   C  : at cOff = (z/cDiv)*sC0 + (z%cDiv)*sC1, element (row,col) at row*ldC+col
//   Res: f32, element ((row % resMod)*N + col)
// Requires M % 64 == 0, N % 32 == 0, K % 32 == 0.
// ---------------------------------------------------------------------------
template <bool BIAS, bool RES, int ACT, bool OBF>
__global__ void k_gemm_wmma(const __bf16* __restrict__ A, const __bf16* __restrict__ BT,
                            void* __restrict__ Cout, const float* __restrict__ bias,
                            const float* __restrict__ Res,
                            int M, int N, int K, float alpha,
                            long long sA, long long sB,
                            int cDiv, long long sC0, long long sC1, int ldC,
                            int resMod) {
    int wave = threadIdx.x >> 5;
    int lane = threadIdx.x & 31;
    int tilesN = N >> 5;                       // 32-col super-tiles
    int tilesTotal = (M >> 6) * tilesN;        // 64-row super-tiles
    int tile = blockIdx.x * (blockDim.x >> 5) + wave;
    if (tile >= tilesTotal) return;            // wave-uniform exit; EXEC all-ones for WMMA
    int tm = tile / tilesN, tn = tile % tilesN;
    int z = blockIdx.z;

    // A fragment: lane holds row (lane&15); K-halves start at (lane>>4)*8 and +16
    const __bf16* Abase = A + (long long)z * sA
                        + (long long)(tm * 64 + (lane & 15)) * K + ((lane >> 4) << 3);
    // B fragment: lane holds col (lane&15); K-halves start at (lane>>4)*8 and +16
    const __bf16* Bbase = BT + (long long)z * sB
                        + (long long)(tn * 32 + (lane & 15)) * K + ((lane >> 4) << 3);

    v8f acc[4][2] = {};
    for (int k0 = 0; k0 < K; k0 += 32) {
        v16bf bv[2];
#pragma unroll
        for (int ni = 0; ni < 2; ni++) {
            const __bf16* Bp = Bbase + (long long)ni * 16 * K + k0;
            v8bf b0 = *(const v8bf*)(Bp);
            v8bf b1 = *(const v8bf*)(Bp + 16);
#pragma unroll
            for (int i = 0; i < 8; i++) { bv[ni][i] = b0[i]; bv[ni][i + 8] = b1[i]; }
        }
#pragma unroll
        for (int mi = 0; mi < 4; mi++) {
            const __bf16* Ap = Abase + (long long)mi * 16 * K + k0;
            v8bf a0 = *(const v8bf*)(Ap);
            v8bf a1 = *(const v8bf*)(Ap + 16);
            v16bf av;
#pragma unroll
            for (int i = 0; i < 8; i++) { av[i] = a0[i]; av[i + 8] = a1[i]; }
#pragma unroll
            for (int ni = 0; ni < 2; ni++)
                acc[mi][ni] = __builtin_amdgcn_wmma_f32_16x16x32_bf16(
                    /*neg_a=*/false, av, /*neg_b=*/false, bv[ni],
                    /*c_mod=*/(short)0, acc[mi][ni],
                    /*reuse_a=*/false, /*reuse_b=*/false);
        }
    }

    long long cOff = (long long)(z / cDiv) * sC0 + (long long)(z % cDiv) * sC1;
    int cn   = lane & 15;
    int rsub = (lane >> 4) << 3;
#pragma unroll
    for (int mi = 0; mi < 4; mi++) {
#pragma unroll
        for (int ni = 0; ni < 2; ni++) {
            int col   = tn * 32 + ni * 16 + cn;
            int rbase = tm * 64 + mi * 16 + rsub;
#pragma unroll
            for (int r = 0; r < 8; r++) {
                float v = acc[mi][ni][r] * alpha;
                if (BIAS) v += bias[col];
                int row = rbase + r;
                if (RES) v += Res[(long long)(row % resMod) * N + col];
                if (ACT == 1) v = tanhf(v);
                long long ci = cOff + (long long)row * ldC + col;
                if (OBF) ((__bf16*)Cout)[ci] = f2bf(v);
                else     ((float*)Cout)[ci]  = v;
            }
        }
    }
}

// ---------------------------------------------------------------------------
// Host launcher
// ---------------------------------------------------------------------------
static inline int cdiv_i(long long a, long long b) { return (int)((a + b - 1) / b); }

extern "C" void kernel_launch(void* const* d_in, const int* in_sizes, int n_in,
                              void* d_out, int out_size, void* d_ws, size_t ws_size,
                              hipStream_t stream) {
    (void)in_sizes; (void)n_in; (void)out_size; (void)ws_size;

    const float* volume  = (const float*)d_in[0];
    const float* conv_w  = (const float*)d_in[1];
    const float* conv_b  = (const float*)d_in[2];
    const float* pos     = (const float*)d_in[3];
    const float* ln1w    = (const float*)d_in[4];
    const float* ln1b    = (const float*)d_in[5];
    const float* wqkv    = (const float*)d_in[6];
    const float* wo      = (const float*)d_in[7];
    const float* bo      = (const float*)d_in[8];
    const float* ln2w    = (const float*)d_in[9];
    const float* ln2b    = (const float*)d_in[10];
    const float* w1      = (const float*)d_in[11];
    const float* b1      = (const float*)d_in[12];
    const float* w2      = (const float*)d_in[13];
    const float* b2      = (const float*)d_in[14];
    const float* lnfw    = (const float*)d_in[15];
    const float* lnfb    = (const float*)d_in[16];

    // ---- workspace carve-up ----
    char* wp = (char*)d_ws;
    auto alloc = [&](size_t bytes) { char* p = wp; wp += (bytes + 255) & ~(size_t)255; return p; };

    __bf16* wcT    = (__bf16*)alloc((size_t)BDIM * KPATCH * 2);          // [768,512]
    __bf16* wqkvT  = (__bf16*)alloc((size_t)4 * QKVN * BDIM * 2);        // [L][2304,768]
    __bf16* woT    = (__bf16*)alloc((size_t)4 * BDIM * BDIM * 2);        // [L][768,768]
    __bf16* w1T    = (__bf16*)alloc((size_t)4 * MLPN * BDIM * 2);        // [L][3072,768]
    __bf16* w2T    = (__bf16*)alloc((size_t)4 * BDIM * MLPN * 2);        // [L][768,3072]
    __bf16* patch  = (__bf16*)alloc((size_t)MROWS * KPATCH * 2);         // [1024,512]
    float*  x      = (float*) alloc((size_t)MROWS * BDIM * 4);           // [1024,768]
    __bf16* hbf    = (__bf16*)alloc((size_t)MROWS * BDIM * 2);           // [1024,768]
    __bf16* qkvbf  = (__bf16*)alloc((size_t)MROWS * QKVN * 2);           // [1024,2304]
    __bf16* qbf    = (__bf16*)alloc((size_t)NBH * NTOK * DHEAD * 2);     // [24,512,64]
    __bf16* kbf    = (__bf16*)alloc((size_t)NBH * NTOK * DHEAD * 2);
    __bf16* vTbf   = (__bf16*)alloc((size_t)NBH * DHEAD * NTOK * 2);     // [24,64,512]
    float*  S      = (float*) alloc((size_t)NBH * NTOK * NTOK * 4);      // [24,512,512]
    __bf16* P      = (__bf16*)alloc((size_t)NBH * NTOK * NTOK * 2);
    __bf16* obf    = (__bf16*)alloc((size_t)MROWS * BDIM * 2);           // [1024,768]
    __bf16* ubf    = (__bf16*)alloc((size_t)MROWS * MLPN * 2);           // [1024,3072]

    // ---- weight prep ----
    k_convert<<<cdiv_i((long long)BDIM * KPATCH, 256), 256, 0, stream>>>(conv_w, wcT, BDIM * KPATCH);
    for (int L = 0; L < 4; L++) {
        k_transpose<<<cdiv_i((long long)BDIM * QKVN, 256), 256, 0, stream>>>(
            wqkv + (long long)L * BDIM * QKVN, wqkvT + (long long)L * QKVN * BDIM, BDIM, QKVN);
        k_transpose<<<cdiv_i((long long)BDIM * BDIM, 256), 256, 0, stream>>>(
            wo + (long long)L * BDIM * BDIM, woT + (long long)L * BDIM * BDIM, BDIM, BDIM);
        k_transpose<<<cdiv_i((long long)BDIM * MLPN, 256), 256, 0, stream>>>(
            w1 + (long long)L * BDIM * MLPN, w1T + (long long)L * MLPN * BDIM, BDIM, MLPN);
        k_transpose<<<cdiv_i((long long)MLPN * BDIM, 256), 256, 0, stream>>>(
            w2 + (long long)L * MLPN * BDIM, w2T + (long long)L * BDIM * MLPN, MLPN, BDIM);
    }

    // ---- patch embedding: x = patch @ wcT^T + conv_b + pos_embed ----
    k_im2col<<<cdiv_i((long long)MROWS * KPATCH, 256), 256, 0, stream>>>(volume, patch);
    {
        int tiles = (MROWS / 64) * (BDIM / 32);
        dim3 g(cdiv_i(tiles, 8), 1, 1);
        k_gemm_wmma<true, true, 0, false><<<g, 256, 0, stream>>>(
            patch, wcT, x, conv_b, pos,
            MROWS, BDIM, KPATCH, 1.0f, 0, 0, 1, 0, 0, BDIM, NTOK);
    }

    const float scale = 0.125f;  // 1/sqrt(64)

    for (int L = 0; L < 4; L++) {
        // LN1 -> h (bf16)
        k_layernorm<true><<<MROWS / 8, 256, 0, stream>>>(
            x, ln1w + L * BDIM, ln1b + L * BDIM, hbf, MROWS, BDIM);

        // QKV = h @ wqkv  (no bias)
        {
            int tiles = (MROWS / 64) * (QKVN / 32);
            dim3 g(cdiv_i(tiles, 8), 1, 1);
            k_gemm_wmma<false, false, 0, true><<<g, 256, 0, stream>>>(
                hbf, wqkvT + (long long)L * QKVN * BDIM, qkvbf, nullptr, nullptr,
                MROWS, QKVN, BDIM, 1.0f, 0, 0, 1, 0, 0, QKVN, 1);
        }
        k_qkv_split<<<cdiv_i((long long)MROWS * QKVN, 256), 256, 0, stream>>>(qkvbf, qbf, kbf, vTbf);

        // S = scale * Q @ K^T per (b,h)
        {
            int tiles = (NTOK / 64) * (NTOK / 32);
            dim3 g(cdiv_i(tiles, 8), 1, NBH);
            k_gemm_wmma<false, false, 0, false><<<g, 256, 0, stream>>>(
                qbf, kbf, S, nullptr, nullptr,
                NTOK, NTOK, DHEAD, scale,
                (long long)NTOK * DHEAD, (long long)NTOK * DHEAD,
                1, (long long)NTOK * NTOK, 0, NTOK, 1);
        }

        // softmax rows -> P (bf16)
        k_softmax<<<(NBH * NTOK) / 8, 256, 0, stream>>>(S, P, NBH * NTOK, NTOK);

        // O = P @ V per (b,h); scatter into obf [token, h*64+d]
        {
            int tiles = (NTOK / 64) * (DHEAD / 32);
            dim3 g(cdiv_i(tiles, 8), 1, NBH);
            k_gemm_wmma<false, false, 0, true><<<g, 256, 0, stream>>>(
                P, vTbf, obf, nullptr, nullptr,
                NTOK, DHEAD, NTOK, 1.0f,
                (long long)NTOK * NTOK, (long long)DHEAD * NTOK,
                NHEAD, (long long)NTOK * BDIM, DHEAD, BDIM, 1);
        }

        // x = x + O @ w_o + b_o
        {
            int tiles = (MROWS / 64) * (BDIM / 32);
            dim3 g(cdiv_i(tiles, 8), 1, 1);
            k_gemm_wmma<true, true, 0, false><<<g, 256, 0, stream>>>(
                obf, woT + (long long)L * BDIM * BDIM, x, bo + L * BDIM, x,
                MROWS, BDIM, BDIM, 1.0f, 0, 0, 1, 0, 0, BDIM, MROWS);
        }

        // LN2 -> h (bf16)
        k_layernorm<true><<<MROWS / 8, 256, 0, stream>>>(
            x, ln2w + L * BDIM, ln2b + L * BDIM, hbf, MROWS, BDIM);

        // u = tanh(h @ W1 + B1)
        {
            int tiles = (MROWS / 64) * (MLPN / 32);
            dim3 g(cdiv_i(tiles, 8), 1, 1);
            k_gemm_wmma<true, false, 1, true><<<g, 256, 0, stream>>>(
                hbf, w1T + (long long)L * MLPN * BDIM, ubf, b1 + L * MLPN, nullptr,
                MROWS, MLPN, BDIM, 1.0f, 0, 0, 1, 0, 0, MLPN, 1);
        }

        // x = x + u @ W2 + B2
        {
            int tiles = (MROWS / 64) * (BDIM / 32);
            dim3 g(cdiv_i(tiles, 8), 1, 1);
            k_gemm_wmma<true, true, 0, false><<<g, 256, 0, stream>>>(
                ubf, w2T + (long long)L * BDIM * MLPN, x, b2 + L * BDIM, x,
                MROWS, BDIM, MLPN, 1.0f, 0, 0, 1, 0, 0, BDIM, MROWS);
        }
    }

    // final LN -> d_out (f32)
    k_layernorm<false><<<MROWS / 8, 256, 0, stream>>>(x, lnfw, lnfb, d_out, MROWS, BDIM);
}
